// CrossFusion_83726092468623
// MI455X (gfx1250) — matrix operand order repaired and checked by reading, not compiled, not run on hardware
//
#include <hip/hip_runtime.h>
#include <hip/hip_bf16.h>

// ---------------------------------------------------------------------------
// CDNA5 (gfx1250) implementation: bf16 WMMA GEMMs + fp32 VALU epilogues.
// GEMM tile fill uses async global->LDS copies (ASYNCcnt) with LDS double
// buffering when the toolchain exposes the builtins; otherwise falls back to
// the synchronous VGPR-staged path.
// ---------------------------------------------------------------------------

typedef __bf16 bf16;
typedef __attribute__((ext_vector_type(16))) __bf16 v16bf;
typedef __attribute__((ext_vector_type(8)))  float  v8f;

#define BDIM   512
#define TGT    49
#define SRC    40
#define EMB    768
#define NH     12
#define HD     64
#define FF     3072

#define MQ     (BDIM * TGT)   // 25088
#define MK     (BDIM * SRC)   // 20480

// GEMM tile config
#define BM 128
#define BN 64
#define BK 64
#define PAD 8
#define LDT (BK + PAD)        // 72 bf16 per LDS row (144B, 16B aligned)

// epilogue flags
#define FLG_QUICKGELU 1
#define FLG_BF16OUT   2
#define FLG_RESID     4

// ---- async global->LDS support probe --------------------------------------
#if defined(__has_builtin)
# if __has_builtin(__builtin_amdgcn_global_load_async_to_lds_b128) && \
     __has_builtin(__builtin_amdgcn_s_wait_asynccnt)
#  define USE_ASYNC_LDS 1
# endif
#endif
#ifndef USE_ASYNC_LDS
# define USE_ASYNC_LDS 0
#endif

#if USE_ASYNC_LDS
// Probe-derived prototype: (v4i AS1*, v4i AS3*, imm offset, imm cpol)
typedef int v4i_vs __attribute__((vector_size(16)));
#define ASYNC_CP128(g, l)                                                      \
    __builtin_amdgcn_global_load_async_to_lds_b128(                            \
        (__attribute__((address_space(1))) v4i_vs*)(                           \
            (__attribute__((address_space(1))) const void*)(g)),               \
        (__attribute__((address_space(3))) v4i_vs*)(                           \
            (__attribute__((address_space(3))) void*)(l)),                     \
        0, 0)
#define NBUF 2
#else
#define NBUF 1
#endif

union Frag16 { v16bf v; uint4 u[2]; };

// ---------------------------------------------------------------------------
// Tiled bf16 GEMM:  C[M,N] = A[M,K](bf16) * Wt[N,K](bf16)^T + bias
// Wt is stored N-major (row n holds K contiguous values) so LDS fill is a
// straight copy matching the CDNA5 B-matrix 32x16 bf16 lane layout.
// ---------------------------------------------------------------------------
__global__ __launch_bounds__(256)
void gemm_bf16_wmma_kernel(const bf16* __restrict__ A,
                           const bf16* __restrict__ Wt,
                           const float* __restrict__ bias,
                           const float* __restrict__ resid,
                           float* __restrict__ outF,
                           bf16* __restrict__ outB,
                           int M, int N, int K, int flags)
{
    __shared__ __align__(16) bf16 As[NBUF][BM * LDT];
    __shared__ __align__(16) bf16 Bs[NBUF][BN * LDT];

    const int tid   = threadIdx.x;
    const int lane  = tid & 31;
    const int wid   = tid >> 5;
    const int waveM = wid & 3;          // 4 waves along M
    const int waveN = wid >> 2;         // 2 waves along N
    const int m0    = blockIdx.y * BM;
    const int n0    = blockIdx.x * BN;

    const int lhalf = (lane >> 4) & 1;  // lanes 0-15 vs 16-31
    const int l16   = lane & 15;

    v8f acc[2][2];
#pragma unroll
    for (int i = 0; i < 2; ++i)
#pragma unroll
        for (int j = 0; j < 2; ++j)
#pragma unroll
            for (int r = 0; r < 8; ++r) acc[i][j][r] = 0.0f;

    const int crow = tid >> 3;          // 0..31: row within a 32-row copy pass
    const int ccol = (tid & 7) * 8;     // bf16 column of this thread's uint4

    // one 32-wide K step of WMMA from the given LDS buffers
    auto compute_ks = [&](const bf16* as, const bf16* bs, int ks) {
        Frag16 fa[2], fb[2];
        // A fragment: lane holds row m=l16, K = {0..7,16..23} (lanes 0-15)
        //                                or {8..15,24..31} (lanes 16-31)
#pragma unroll
        for (int mi = 0; mi < 2; ++mi) {
            const int row = waveM * 32 + mi * 16 + l16;
            const int off = row * LDT + ks + (lhalf ? 8 : 0);
            fa[mi].u[0] = *(const uint4*)(&as[off]);
            fa[mi].u[1] = *(const uint4*)(&as[off + 16]);
        }
        // B fragment: lane holds col n=l16, K = {0..7,16..23} (lanes 0-15)
        //                                   or {8..15,24..31} (lanes 16-31)
#pragma unroll
        for (int ni = 0; ni < 2; ++ni) {
            const int row = waveN * 32 + ni * 16 + l16;
            const int off = row * LDT + ks + (lhalf ? 8 : 0);
            fb[ni].u[0] = *(const uint4*)(&bs[off]);
            fb[ni].u[1] = *(const uint4*)(&bs[off + 16]);
        }
#pragma unroll
        for (int mi = 0; mi < 2; ++mi)
#pragma unroll
            for (int ni = 0; ni < 2; ++ni)
                acc[mi][ni] = __builtin_amdgcn_wmma_f32_16x16x32_bf16(
                    false, fa[mi].v, false, fb[ni].v,
                    (short)0, acc[mi][ni], false, false);
    };

#if USE_ASYNC_LDS
    // ---- async double-buffered pipeline (6 async b128 copies per stage) ----
    auto issue_copy = [&](int kb, int buf) {
#pragma unroll
        for (int p = 0; p < 4; ++p) {
            const int r = crow + p * 32;
            ASYNC_CP128(A + (size_t)(m0 + r) * K + kb + ccol,
                        &As[buf][r * LDT + ccol]);
        }
#pragma unroll
        for (int p = 0; p < 2; ++p) {
            const int r = crow + p * 32;
            ASYNC_CP128(Wt + (size_t)(n0 + r) * K + kb + ccol,
                        &Bs[buf][r * LDT + ccol]);
        }
    };

    const int nk = K / BK;
    issue_copy(0, 0);
    int buf = 0;
    for (int it = 0; it < nk; ++it) {
        if (it + 1 < nk) {
            // prefetch next tile into the other buffer (last read in the previous stage,
            // protected by the trailing barrier of that stage)
            issue_copy((it + 1) * BK, buf ^ 1);
            __builtin_amdgcn_s_wait_asynccnt(6);  // oldest 6 (current buf) done
        } else {
            __builtin_amdgcn_s_wait_asynccnt(0);
        }
        __syncthreads();
#pragma unroll
        for (int ks = 0; ks < BK; ks += 32)
            compute_ks(&As[buf][0], &Bs[buf][0], ks);
        __syncthreads();
        buf ^= 1;
    }
#else
    // ---- synchronous fallback: VGPR-staged copies, single buffer ----------
    for (int kb = 0; kb < K; kb += BK) {
#pragma unroll
        for (int p = 0; p < 4; ++p) {
            const int r = crow + p * 32;
            const uint4 g = *(const uint4*)(A + (size_t)(m0 + r) * K + kb + ccol);
            *(uint4*)(&As[0][r * LDT + ccol]) = g;
        }
#pragma unroll
        for (int p = 0; p < 2; ++p) {
            const int r = crow + p * 32;
            const uint4 g = *(const uint4*)(Wt + (size_t)(n0 + r) * K + kb + ccol);
            *(uint4*)(&Bs[0][r * LDT + ccol]) = g;
        }
        __syncthreads();
#pragma unroll
        for (int ks = 0; ks < BK; ks += 32)
            compute_ks(&As[0][0], &Bs[0][0], ks);
        __syncthreads();
    }
#endif

    // ---- epilogue: C/D layout => row m = vgpr + (lane>=16 ? 8 : 0), col = l16
#pragma unroll
    for (int mi = 0; mi < 2; ++mi) {
#pragma unroll
        for (int ni = 0; ni < 2; ++ni) {
#pragma unroll
            for (int r = 0; r < 8; ++r) {
                const int m = m0 + waveM * 32 + mi * 16 + r + lhalf * 8;
                const int n = n0 + waveN * 32 + ni * 16 + l16;
                const size_t idx = (size_t)m * N + n;
                float c = acc[mi][ni][r] + bias[n];
                if (flags & FLG_RESID)      c += resid[idx];
                if (flags & FLG_QUICKGELU)  c = c / (1.0f + __expf(-1.702f * c));
                if (flags & FLG_BF16OUT)    outB[idx] = (bf16)c;
                else                        outF[idx] = c;
            }
        }
    }
}

// ---------------------------------------------------------------------------
// fp32 -> bf16 convert (activations)
// ---------------------------------------------------------------------------
__global__ void cvt_bf16_kernel(const float* __restrict__ s, bf16* __restrict__ d, int n)
{
    int i = blockIdx.x * 256 + threadIdx.x;
    if (i < n) d[i] = (bf16)s[i];
}

// fp32 W[K,N] -> bf16 Wt[N,K]
__global__ void transpose_cvt_kernel(const float* __restrict__ W, bf16* __restrict__ Wt,
                                     int K, int N)
{
    int i = blockIdx.x * 256 + threadIdx.x;
    if (i < K * N) {
        int k = i / N, n = i % N;
        Wt[(size_t)n * K + k] = (bf16)W[i];
    }
}

// ---------------------------------------------------------------------------
// LayerNorm over E=768, one block (256 thr) per row, bf16 output.
// ---------------------------------------------------------------------------
__global__ __launch_bounds__(256)
void layernorm_bf16_kernel(const float* __restrict__ x, const float* __restrict__ g,
                           const float* __restrict__ b, bf16* __restrict__ y)
{
    const int row = blockIdx.x;
    const int t = threadIdx.x;
    const float* xr = x + (size_t)row * EMB;
    float v0 = xr[t], v1 = xr[t + 256], v2 = xr[t + 512];
    __shared__ float rs[256], rss[256];
    rs[t]  = v0 + v1 + v2;
    rss[t] = v0 * v0 + v1 * v1 + v2 * v2;
    __syncthreads();
    for (int o = 128; o > 0; o >>= 1) {
        if (t < o) { rs[t] += rs[t + o]; rss[t] += rss[t + o]; }
        __syncthreads();
    }
    const float mu   = rs[0] * (1.0f / EMB);
    const float var  = rss[0] * (1.0f / EMB) - mu * mu;
    const float rstd = rsqrtf(var + 1e-5f);
    bf16* yr = y + (size_t)row * EMB;
    yr[t]       = (bf16)((v0 - mu) * rstd * g[t]       + b[t]);
    yr[t + 256] = (bf16)((v1 - mu) * rstd * g[t + 256] + b[t + 256]);
    yr[t + 512] = (bf16)((v2 - mu) * rstd * g[t + 512] + b[t + 512]);
}

// ---------------------------------------------------------------------------
// q path: q[B,49,768] -> (over seq) trs (49->40) + exact GELU -> qt[B,40,768]
// one thread per (b, e) column.
// ---------------------------------------------------------------------------
__global__ __launch_bounds__(256)
void q_trs_gelu_kernel(const float* __restrict__ q, const float* __restrict__ trs_w,
                       const float* __restrict__ trs_b, float* __restrict__ qt)
{
    __shared__ float w[TGT * SRC];
    __shared__ float tb[SRC];
    for (int i = threadIdx.x; i < TGT * SRC; i += 256) w[i] = trs_w[i];
    if (threadIdx.x < SRC) tb[threadIdx.x] = trs_b[threadIdx.x];
    __syncthreads();

    const int b = blockIdx.x / 3;
    const int e = (blockIdx.x % 3) * 256 + threadIdx.x;
    const float* qb = q + (size_t)b * TGT * EMB + e;
    float acc[SRC];
#pragma unroll
    for (int j = 0; j < SRC; ++j) acc[j] = tb[j];
    for (int i = 0; i < TGT; ++i) {
        const float qv = qb[(size_t)i * EMB];
#pragma unroll
        for (int j = 0; j < SRC; ++j) acc[j] += qv * w[i * SRC + j];
    }
    float* qo = qt + (size_t)b * SRC * EMB + e;
#pragma unroll
    for (int j = 0; j < SRC; ++j) {
        const float xx = acc[j];
        qo[(size_t)j * EMB] = 0.5f * xx * (1.0f + erff(xx * 0.70710678118f));
    }
}

// ---------------------------------------------------------------------------
// Per-head: Bk = (q@wqk+wqk_b)*(k@wk+wk_b) ; bscore = Bk@wb + wb_b
// grid = B*H*SRC blocks, 64 threads. The raw reshape (B,H,40,64) of a
// [B,40,768] tensor is a flat re-index: base = b*30720 + h*2560 + s*64.
// ---------------------------------------------------------------------------
__global__ __launch_bounds__(64)
void head_bk_kernel(const float* __restrict__ qt, const float* __restrict__ kk,
                    const float* __restrict__ wqk_w, const float* __restrict__ wqk_b,
                    const float* __restrict__ wk_w,  const float* __restrict__ wk_b,
                    const float* __restrict__ wb_w,  const float* __restrict__ wb_b,
                    float* __restrict__ Bk, float* __restrict__ bscore)
{
    const int r = blockIdx.x;
    const int d = threadIdx.x;
    const int s = r % SRC;
    const int h = (r / SRC) % NH;
    const int b = r / (SRC * NH);
    const size_t base = (size_t)b * (NH * SRC * HD) + (size_t)h * (SRC * HD) + (size_t)s * HD;

    __shared__ float qs[HD], ks[HD], red[HD];
    qs[d] = qt[base + d];
    ks[d] = kk[base + d];
    __syncthreads();

    float a1 = wqk_b[d], a2 = wk_b[d];
    for (int t = 0; t < HD; ++t) {
        a1 += qs[t] * wqk_w[t * HD + d];
        a2 += ks[t] * wk_w[t * HD + d];
    }
    const float bk = a1 * a2;
    Bk[(size_t)r * HD + d] = bk;

    red[d] = bk * wb_w[d];
    __syncthreads();
    for (int o = 32; o > 0; o >>= 1) {
        if (d < o) red[d] += red[d + o];
        __syncthreads();
    }
    if (d == 0) bscore[r] = red[0] + wb_b[0];
}

// ---------------------------------------------------------------------------
// Per (b,h): m[d] = sum_s Bk[s,d]*wm_w[s] + wm_b ; softmax(m over 64) - 0.05
//            softmax(bscore over 40)
// ---------------------------------------------------------------------------
__global__ __launch_bounds__(64)
void head_softmax_kernel(const float* __restrict__ Bk, const float* __restrict__ wm_w,
                         const float* __restrict__ wm_b, const float* __restrict__ bscore,
                         float* __restrict__ attn1, float* __restrict__ attn2)
{
    const int g = blockIdx.x;    // b*NH + h
    const int d = threadIdx.x;
    __shared__ float sm[HD];

    // ---- attn2 : softmax over head dim (64)
    float m = wm_b[0];
    const size_t base = (size_t)g * (SRC * HD);
    for (int s = 0; s < SRC; ++s) m += Bk[base + s * HD + d] * wm_w[s];
    sm[d] = m; __syncthreads();
    for (int o = 32; o > 0; o >>= 1) { if (d < o) sm[d] = fmaxf(sm[d], sm[d + o]); __syncthreads(); }
    const float mx = sm[0]; __syncthreads();
    const float e = __expf(m - mx);
    sm[d] = e; __syncthreads();
    for (int o = 32; o > 0; o >>= 1) { if (d < o) sm[d] += sm[d + o]; __syncthreads(); }
    attn2[(size_t)g * HD + d] = e / sm[0] - 0.05f;
    __syncthreads();

    // ---- attn1 : softmax over the 40 (src) dim
    const float bs = (d < SRC) ? bscore[(size_t)g * SRC + d] : -3.0e38f;
    sm[d] = bs; __syncthreads();
    for (int o = 32; o > 0; o >>= 1) { if (d < o) sm[d] = fmaxf(sm[d], sm[d + o]); __syncthreads(); }
    const float mx2 = sm[0]; __syncthreads();
    const float e2 = (d < SRC) ? __expf(bs - mx2) : 0.0f;
    sm[d] = e2; __syncthreads();
    for (int o = 32; o > 0; o >>= 1) { if (d < o) sm[d] += sm[d + o]; __syncthreads(); }
    if (d < SRC) attn1[(size_t)g * SRC + d] = e2 / sm[0];
}

// ---------------------------------------------------------------------------
// Per (b,h): w[s,d]=(attn1[s]+attn2[d])*v[s,d]; out2[t,d]=sum_s w[s,d]*trs2[s,t]
// written directly in [B,49,768] bf16 layout (head transpose folded in).
// ---------------------------------------------------------------------------
__global__ __launch_bounds__(256)
void head_av_trs2_kernel(const float* __restrict__ v, const float* __restrict__ attn1,
                         const float* __restrict__ attn2, const float* __restrict__ trs2_w,
                         const float* __restrict__ trs2_b, bf16* __restrict__ y)
{
    const int g = blockIdx.x;          // b*NH + h
    const int h = g % NH;
    const int b = g / NH;
    __shared__ float w[SRC * HD];
    __shared__ float a1s[SRC], a2s[HD], tb[TGT];

    const int t = threadIdx.x;
    if (t < SRC) a1s[t] = attn1[(size_t)g * SRC + t];
    if (t < HD)  a2s[t] = attn2[(size_t)g * HD + t];
    if (t < TGT) tb[t]  = trs2_b[t];
    __syncthreads();

    const size_t vbase = (size_t)b * (NH * SRC * HD) + (size_t)h * (SRC * HD);
    for (int i = t; i < SRC * HD; i += 256) {
        const int s = i >> 6, d = i & 63;
        w[i] = (a1s[s] + a2s[d]) * v[vbase + i];
    }
    __syncthreads();

    for (int o = t; o < TGT * HD; o += 256) {
        const int tt = o >> 6, d = o & 63;
        float acc = tb[tt];
        for (int s = 0; s < SRC; ++s) acc += w[s * HD + d] * trs2_w[s * TGT + tt];
        y[((size_t)b * TGT + tt) * EMB + h * HD + d] = (bf16)acc;
    }
}

// ---------------------------------------------------------------------------
// Host launcher
// ---------------------------------------------------------------------------
static inline size_t alignup(size_t x) { return (x + 255) & ~(size_t)255; }

extern "C" void kernel_launch(void* const* d_in, const int* in_sizes, int n_in,
                              void* d_out, int out_size, void* d_ws, size_t ws_size,
                              hipStream_t stream)
{
    const float* hidden = (const float*)d_in[0];
    const float* pkv    = (const float*)d_in[1];
    const float* ln1_g  = (const float*)d_in[2];
    const float* ln1_b  = (const float*)d_in[3];
    const float* q_w    = (const float*)d_in[4];
    const float* q_b    = (const float*)d_in[5];
    const float* k_w    = (const float*)d_in[6];
    const float* k_b    = (const float*)d_in[7];
    const float* v_w    = (const float*)d_in[8];
    const float* v_b    = (const float*)d_in[9];
    const float* wqk_w  = (const float*)d_in[10];
    const float* wqk_b  = (const float*)d_in[11];
    const float* wk_w   = (const float*)d_in[12];
    const float* wk_b   = (const float*)d_in[13];
    const float* wb_w   = (const float*)d_in[14];
    const float* wb_b   = (const float*)d_in[15];
    const float* wm_w   = (const float*)d_in[16];
    const float* wm_b   = (const float*)d_in[17];
    const float* trs_w  = (const float*)d_in[18];
    const float* trs_b  = (const float*)d_in[19];
    const float* trs2_w = (const float*)d_in[20];
    const float* trs2_b = (const float*)d_in[21];
    const float* out_w  = (const float*)d_in[22];
    const float* out_b  = (const float*)d_in[23];
    const float* ln2_g  = (const float*)d_in[24];
    const float* ln2_b  = (const float*)d_in[25];
    const float* fc1_w  = (const float*)d_in[26];
    const float* fc1_b  = (const float*)d_in[27];
    const float* fc2_w  = (const float*)d_in[28];
    const float* fc2_b  = (const float*)d_in[29];

    char* base = (char*)d_ws;
    size_t off = 0;
    auto take = [&](size_t bytes) { char* p = base + off; off += alignup(bytes); return p; };

    bf16* wq_t   = (bf16*)take((size_t)EMB * EMB * 2);
    bf16* wk_t   = (bf16*)take((size_t)EMB * EMB * 2);
    bf16* wv_t   = (bf16*)take((size_t)EMB * EMB * 2);
    bf16* wout_t = (bf16*)take((size_t)EMB * EMB * 2);
    bf16* wfc1_t = (bf16*)take((size_t)FF  * EMB * 2);  // [3072][768]
    bf16* wfc2_t = (bf16*)take((size_t)EMB * FF  * 2);  // [768][3072]
    bf16* pkv_bf = (bf16*)take((size_t)MK  * EMB * 2);
    bf16* xln    = (bf16*)take((size_t)MQ  * EMB * 2);  // reused later as ln2 output
    float* qbuf  = (float*)take((size_t)MQ * EMB * 4);  // reused later as hs
    char*  bigrg = (char*)take((size_t)3 * MK * EMB * 4); // qt | k | v ; later fc1 act
    float* qt    = (float*)bigrg;
    float* kbuf  = (float*)(bigrg + (size_t)MK * EMB * 4);
    float* vbuf  = (float*)(bigrg + (size_t)2 * MK * EMB * 4);
    bf16*  fc1a  = (bf16*)bigrg;                        // 154 MB <= 189 MB region
    float* Bk    = (float*)take((size_t)MK * EMB * 4);  // [B,H,40,64] fp32
    bf16*  ybuf  = (bf16*)Bk;                           // y aliases Bk (Bk dead by then)
    float* bscore = (float*)take((size_t)BDIM * NH * SRC * 4);
    float* attn1  = (float*)take((size_t)BDIM * NH * SRC * 4);
    float* attn2  = (float*)take((size_t)BDIM * NH * HD * 4);
    (void)ws_size; (void)n_in; (void)in_sizes; (void)out_size;

    float* hs  = qbuf;   // residual-1 output aliases q (q dead after trs)
    bf16*  ln2 = xln;    // ln2 output aliases xln (dead after q GEMM)
    float* outp = (float*)d_out;

    auto gemm = [&](const bf16* A, const bf16* Wt, const float* bias, const float* resid,
                    float* oF, bf16* oB, int M, int N, int K, int flags) {
        dim3 grid(N / BN, M / BM);
        gemm_bf16_wmma_kernel<<<grid, dim3(256), 0, stream>>>(A, Wt, bias, resid, oF, oB,
                                                              M, N, K, flags);
    };

    // 1) weight convert+transpose (once per launch; weights are tiny)
    {
        int n = EMB * EMB;
        transpose_cvt_kernel<<<(n + 255) / 256, 256, 0, stream>>>(q_w,   wq_t,   EMB, EMB);
        transpose_cvt_kernel<<<(n + 255) / 256, 256, 0, stream>>>(k_w,   wk_t,   EMB, EMB);
        transpose_cvt_kernel<<<(n + 255) / 256, 256, 0, stream>>>(v_w,   wv_t,   EMB, EMB);
        transpose_cvt_kernel<<<(n + 255) / 256, 256, 0, stream>>>(out_w, wout_t, EMB, EMB);
        int nf = EMB * FF;
        transpose_cvt_kernel<<<(nf + 255) / 256, 256, 0, stream>>>(fc1_w, wfc1_t, EMB, FF);
        transpose_cvt_kernel<<<(nf + 255) / 256, 256, 0, stream>>>(fc2_w, wfc2_t, FF, EMB);
    }
    // 2) past_key_values -> bf16
    {
        int n = MK * EMB;
        cvt_bf16_kernel<<<(n + 255) / 256, 256, 0, stream>>>(pkv, pkv_bf, n);
    }
    // 3) LN1 -> bf16
    layernorm_bf16_kernel<<<MQ, 256, 0, stream>>>(hidden, ln1_g, ln1_b, xln);

    // 4-6) q/k/v projections (bf16 WMMA, fp32 out)
    gemm(xln,    wq_t, q_b, nullptr, qbuf, nullptr, MQ, EMB, EMB, 0);
    gemm(pkv_bf, wk_t, k_b, nullptr, kbuf, nullptr, MK, EMB, EMB, 0);
    gemm(pkv_bf, wv_t, v_b, nullptr, vbuf, nullptr, MK, EMB, EMB, 0);

    // 7) trs (49->40) + exact GELU along seq dim
    q_trs_gelu_kernel<<<BDIM * 3, 256, 0, stream>>>(qbuf, trs_w, trs_b, qt);

    // 8) per-head Bk and b-score
    head_bk_kernel<<<BDIM * NH * SRC, 64, 0, stream>>>(qt, kbuf, wqk_w, wqk_b, wk_w, wk_b,
                                                       wb_w, wb_b, Bk, bscore);
    // 9) dual softmax
    head_softmax_kernel<<<BDIM * NH, 64, 0, stream>>>(Bk, wm_w, wm_b, bscore, attn1, attn2);

    // 10) (attn1+attn2)*v, trs2 (40->49), write [B,49,768] bf16
    head_av_trs2_kernel<<<BDIM * NH, 256, 0, stream>>>(vbuf, attn1, attn2, trs2_w, trs2_b, ybuf);

    // 11) out projection + residual(hidden) -> hs (fp32)
    gemm(ybuf, wout_t, out_b, hidden, hs, nullptr, MQ, EMB, EMB, FLG_RESID);

    // 12) LN2 -> bf16
    layernorm_bf16_kernel<<<MQ, 256, 0, stream>>>(hs, ln2_g, ln2_b, ln2);

    // 13) fc1 + quickGELU -> bf16
    gemm(ln2, wfc1_t, fc1_b, nullptr, nullptr, fc1a, MQ, FF, EMB, FLG_QUICKGELU | FLG_BF16OUT);

    // 14) fc2 + residual(hs) -> d_out (fp32)
    gemm(fc1a, wfc2_t, fc2_b, hs, outp, nullptr, MQ, EMB, FF, FLG_RESID);
}
